// BitnetMLP_3453153706638
// MI455X (gfx1250) — 8-bit matrix operand order repaired and checked by reading, compiled, not run on hardware
//
#include <hip/hip_runtime.h>
#include <hip/hip_bf16.h>

typedef int   v8i __attribute__((ext_vector_type(8)));
typedef float v4f __attribute__((ext_vector_type(4)));

#define TOK  3072
#define DD   2048
#define FF   8192
#define WN   16777216
#define NCLS 1000

__device__ __forceinline__ void async_ld16(void* lds, const void* g) {
  asm volatile("global_load_async_to_lds_b128 %0, %1, off"
               :: "v"((unsigned)(unsigned long long)lds),
                  "v"((unsigned long long)g)
               : "memory");
}
__device__ __forceinline__ void async_wait0() {
  asm volatile("s_wait_asynccnt 0x0" ::: "memory");
}

__global__ void k_wabs_part(const float* __restrict__ Wg, const float* __restrict__ Wu,
                            const float* __restrict__ Wd, float* __restrict__ wpart) {
  const int w  = blockIdx.x >> 10;
  const int bi = blockIdx.x & 1023;
  const float* src = (w == 0) ? Wg : ((w == 1) ? Wu : Wd);
  int base = bi * 16384 + threadIdx.x;
  float s = 0.f;
#pragma unroll 4
  for (int i = 0; i < 64; ++i) s += fabsf(__builtin_nontemporal_load(src + base + i * 256));
  __shared__ float red[256];
  red[threadIdx.x] = s; __syncthreads();
  for (int off = 128; off; off >>= 1) {
    if ((int)threadIdx.x < off) red[threadIdx.x] += red[threadIdx.x + off];
    __syncthreads();
  }
  if (threadIdx.x == 0) wpart[(w << 10) + bi] = red[0];
}

__global__ void k_wscale(const float* __restrict__ wpart, float* __restrict__ winv) {
  const int w = blockIdx.x;
  __shared__ float red[256];
  float s = 0.f;
  for (int i = threadIdx.x; i < 1024; i += 256) s += wpart[(w << 10) + i];
  red[threadIdx.x] = s; __syncthreads();
  for (int off = 128; off; off >>= 1) {
    if ((int)threadIdx.x < off) red[threadIdx.x] += red[threadIdx.x + off];
    __syncthreads();
  }
  if (threadIdx.x == 0) winv[w] = fmaxf(red[0] * (1.0f / 16777216.0f), 1e-5f);
}

__global__ void k_wquant(const float* __restrict__ Wg, const float* __restrict__ Wu,
                         const float* __restrict__ Wd, const float* __restrict__ winv,
                         signed char* __restrict__ wq) {
  int i = (blockIdx.x * 256 + threadIdx.x) * 4;
  int w = i >> 24;
  const float* src = (w == 0) ? Wg : ((w == 1) ? Wu : Wd);
  int off = i & (WN - 1);
  v4f v = __builtin_nontemporal_load((const v4f*)(src + off));
  float s = 1.0f / winv[w];
  float qx = fminf(fmaxf(rintf(v.x * s), -1.f), 1.f);
  float qy = fminf(fmaxf(rintf(v.y * s), -1.f), 1.f);
  float qz = fminf(fmaxf(rintf(v.z * s), -1.f), 1.f);
  float qw = fminf(fmaxf(rintf(v.w * s), -1.f), 1.f);
  char4 o; o.x = (signed char)qx; o.y = (signed char)qy; o.z = (signed char)qz; o.w = (signed char)qw;
  *(char4*)(wq + i) = o;
}

__global__ void k_actq(const float* __restrict__ x, signed char* __restrict__ xq,
                       float* __restrict__ xinv) {
  const int t = blockIdx.x;
  const float* row = x + t * DD;
  float v[8], m = 0.f;
#pragma unroll
  for (int i = 0; i < 8; ++i) { v[i] = row[threadIdx.x + i * 256]; m = fmaxf(m, fabsf(v[i])); }
  __shared__ float red[256];
  red[threadIdx.x] = m; __syncthreads();
  for (int off = 128; off; off >>= 1) {
    if ((int)threadIdx.x < off) red[threadIdx.x] = fmaxf(red[threadIdx.x], red[threadIdx.x + off]);
    __syncthreads();
  }
  float amax  = fmaxf(red[0], 1e-5f);
  float scale = 127.0f / amax;
#pragma unroll
  for (int i = 0; i < 8; ++i) {
    float q = fminf(fmaxf(rintf(v[i] * scale), -128.f), 127.f);
    xq[t * DD + threadIdx.x + i * 256] = (signed char)q;
  }
  if (threadIdx.x == 0) xinv[t] = amax * (1.0f / 127.0f);
}

__global__ __launch_bounds__(256) void k_gemm_gu(
    const signed char* __restrict__ xq, const signed char* __restrict__ wqg,
    const signed char* __restrict__ wqu, const float* __restrict__ xinv,
    const float* __restrict__ winv, float* __restrict__ tbuf) {
  __shared__ __align__(16) signed char As[2][128 * 64];
  __shared__ __align__(16) signed char Bg[2][128 * 64];
  __shared__ __align__(16) signed char Bu[2][128 * 64];
  const int tid = threadIdx.x, lane = tid & 31, wave = tid >> 5;
  const int m0 = blockIdx.x * 128, n0 = blockIdx.y * 128;
  const int lr = tid >> 2, lc = (tid & 3) * 16;
  const int wm = (wave >> 1) * 32, wn = (wave & 1) * 64;
  const int half = lane >> 4, l15 = lane & 15;

  const signed char* gA = xq  + (m0 + lr) * DD + lc;
  const signed char* gG = wqg + (n0 + lr) * DD + lc;
  const signed char* gU = wqu + (n0 + lr) * DD + lc;

  v8i zero = {0, 0, 0, 0, 0, 0, 0, 0};
  v8i accG[2][4], accU[2][4];
#pragma unroll
  for (int mi = 0; mi < 2; ++mi)
#pragma unroll
    for (int j = 0; j < 4; ++j) { accG[mi][j] = zero; accU[mi][j] = zero; }

  auto load_tile = [&](int buf, int k0) {
    async_ld16(&As[buf][lr * 64 + lc],        gA + k0);
    async_ld16(&As[buf][(lr + 64) * 64 + lc], gA + 64 * DD + k0);
    async_ld16(&Bg[buf][lr * 64 + lc],        gG + k0);
    async_ld16(&Bg[buf][(lr + 64) * 64 + lc], gG + 64 * DD + k0);
    async_ld16(&Bu[buf][lr * 64 + lc],        gU + k0);
    async_ld16(&Bu[buf][(lr + 64) * 64 + lc], gU + 64 * DD + k0);
  };

  load_tile(0, 0);
  async_wait0();
  __syncthreads();

  for (int k0 = 0; k0 < DD; k0 += 64) {
    const int cur = (k0 >> 6) & 1;
    if (k0 + 64 < DD) load_tile(cur ^ 1, k0 + 64);
    if (k0 + 128 < DD) {
      __builtin_prefetch(gG + 64 * DD + k0 + 128);
      __builtin_prefetch(gU + 64 * DD + k0 + 128);
    }

    v8i af[2];
#pragma unroll
    for (int mi = 0; mi < 2; ++mi)
#pragma unroll
      for (int p = 0; p < 4; ++p) {
        int2 q = *(const int2*)&As[cur][(wm + mi * 16 + l15) * 64 + (p >> 1) * 32 + (p & 1) * 8 + half * 16];
        af[mi][2 * p] = q.x; af[mi][2 * p + 1] = q.y;
      }
#pragma unroll
    for (int j = 0; j < 4; ++j) {
      const int colL = wn + j * 16 + l15;
      int4 c0 = *(const int4*)&Bg[cur][colL * 64 + half * 16];
      int4 c1 = *(const int4*)&Bg[cur][colL * 64 + half * 16 + 32];
      v8i bfG = {c0.x, c0.y, c0.z, c0.w, c1.x, c1.y, c1.z, c1.w};
      int4 d0 = *(const int4*)&Bu[cur][colL * 64 + half * 16];
      int4 d1 = *(const int4*)&Bu[cur][colL * 64 + half * 16 + 32];
      v8i bfU = {d0.x, d0.y, d0.z, d0.w, d1.x, d1.y, d1.z, d1.w};
#pragma unroll
      for (int mi = 0; mi < 2; ++mi) {
        accG[mi][j] = __builtin_amdgcn_wmma_i32_16x16x64_iu8(true, af[mi], true, bfG, accG[mi][j], false, false);
        accU[mi][j] = __builtin_amdgcn_wmma_i32_16x16x64_iu8(true, af[mi], true, bfU, accU[mi][j], false, false);
      }
    }
    async_wait0();
    __syncthreads();
  }

  const float wg_inv = winv[0], wu_inv = winv[1];
#pragma unroll
  for (int mi = 0; mi < 2; ++mi)
#pragma unroll
    for (int j = 0; j < 4; ++j)
#pragma unroll
      for (int v = 0; v < 8; ++v) {
        int row = m0 + wm + mi * 16 + half * 8 + v;
        int col = n0 + wn + j * 16 + l15;
        float dq = xinv[row];
        float g = (float)accG[mi][j][v] * dq * wg_inv;
        float u = (float)accU[mi][j][v] * dq * wu_inv;
        float sg = 1.0f / (1.0f + __expf(-g));
        tbuf[row * FF + col] = g * sg * u;
      }
}

__global__ void k_rms_quant(const float* __restrict__ tbuf, const float* __restrict__ ln_w,
                            signed char* __restrict__ hq, float* __restrict__ hinv) {
  const int t = blockIdx.x;
  const float* row = tbuf + t * FF;
  float ssq = 0.f, am = 0.f;
  for (int i = threadIdx.x; i < FF; i += 256) {
    float v = row[i];
    ssq += v * v;
    am = fmaxf(am, fabsf(v * ln_w[i]));
  }
  __shared__ float r1[256], r2[256];
  r1[threadIdx.x] = ssq; r2[threadIdx.x] = am; __syncthreads();
  for (int off = 128; off; off >>= 1) {
    if ((int)threadIdx.x < off) {
      r1[threadIdx.x] += r1[threadIdx.x + off];
      r2[threadIdx.x] = fmaxf(r2[threadIdx.x], r2[threadIdx.x + off]);
    }
    __syncthreads();
  }
  float rinv  = rsqrtf(r1[0] * (1.0f / (float)FF) + 1e-6f);
  float amax  = fmaxf(r2[0] * rinv, 1e-5f);
  float scale = 127.0f / amax;
  for (int i = threadIdx.x; i < FF; i += 256) {
    float n = row[i] * ln_w[i] * rinv;
    float q = fminf(fmaxf(rintf(n * scale), -128.f), 127.f);
    hq[t * FF + i] = (signed char)q;
  }
  if (threadIdx.x == 0) hinv[t] = amax * (1.0f / 127.0f);
}

__global__ __launch_bounds__(256) void k_gemm_down(
    const signed char* __restrict__ hq, const signed char* __restrict__ wqd,
    const float* __restrict__ hinv, const float* __restrict__ winv,
    float* __restrict__ part) {
  __shared__ __align__(16) signed char As[2][128 * 64];
  __shared__ __align__(16) signed char Bd[2][128 * 64];
  __shared__ float red[8];
  const int tid = threadIdx.x, lane = tid & 31, wave = tid >> 5;
  const int m0 = blockIdx.x * 128, n0 = blockIdx.y * 128;
  const int lr = tid >> 2, lc = (tid & 3) * 16;
  const int wm = (wave >> 1) * 32, wn = (wave & 1) * 64;
  const int half = lane >> 4, l15 = lane & 15;

  const signed char* gA = hq  + (m0 + lr) * FF + lc;
  const signed char* gB = wqd + (n0 + lr) * FF + lc;

  v8i zero = {0, 0, 0, 0, 0, 0, 0, 0};
  v8i acc[2][4];
#pragma unroll
  for (int mi = 0; mi < 2; ++mi)
#pragma unroll
    for (int j = 0; j < 4; ++j) acc[mi][j] = zero;

  auto load_tile = [&](int buf, int k0) {
    async_ld16(&As[buf][lr * 64 + lc],        gA + k0);
    async_ld16(&As[buf][(lr + 64) * 64 + lc], gA + 64 * FF + k0);
    async_ld16(&Bd[buf][lr * 64 + lc],        gB + k0);
    async_ld16(&Bd[buf][(lr + 64) * 64 + lc], gB + 64 * FF + k0);
  };

  load_tile(0, 0);
  async_wait0();
  __syncthreads();

  for (int k0 = 0; k0 < FF; k0 += 64) {
    const int cur = (k0 >> 6) & 1;
    if (k0 + 64 < FF) load_tile(cur ^ 1, k0 + 64);
    if (k0 + 128 < FF) __builtin_prefetch(gB + 64 * FF + k0 + 128);

    v8i af[2];
#pragma unroll
    for (int mi = 0; mi < 2; ++mi)
#pragma unroll
      for (int p = 0; p < 4; ++p) {
        int2 q = *(const int2*)&As[cur][(wm + mi * 16 + l15) * 64 + (p >> 1) * 32 + (p & 1) * 8 + half * 16];
        af[mi][2 * p] = q.x; af[mi][2 * p + 1] = q.y;
      }
#pragma unroll
    for (int j = 0; j < 4; ++j) {
      const int colL = wn + j * 16 + l15;
      int4 c0 = *(const int4*)&Bd[cur][colL * 64 + half * 16];
      int4 c1 = *(const int4*)&Bd[cur][colL * 64 + half * 16 + 32];
      v8i bf = {c0.x, c0.y, c0.z, c0.w, c1.x, c1.y, c1.z, c1.w};
#pragma unroll
      for (int mi = 0; mi < 2; ++mi)
        acc[mi][j] = __builtin_amdgcn_wmma_i32_16x16x64_iu8(true, af[mi], true, bf, acc[mi][j], false, false);
    }
    async_wait0();
    __syncthreads();
  }

  float lsum = 0.f;
#pragma unroll
  for (int mi = 0; mi < 2; ++mi)
#pragma unroll
    for (int j = 0; j < 4; ++j)
#pragma unroll
      for (int v = 0; v < 8; ++v) {
        int row = m0 + wm + mi * 16 + half * 8 + v;
        lsum += (float)acc[mi][j][v] * hinv[row];
      }
  lsum *= winv[2];
#pragma unroll
  for (int off = 16; off; off >>= 1) lsum += __shfl_xor(lsum, off, 32);
  if (lane == 0) red[wave] = lsum;
  __syncthreads();
  if (tid == 0) {
    float s = 0.f;
#pragma unroll
    for (int w = 0; w < 8; ++w) s += red[w];
    part[blockIdx.x * 16 + blockIdx.y] = s;
  }
}

__global__ void k_pool(const float* __restrict__ part, float* __restrict__ pooled) {
  int i = threadIdx.x;
  if (i < 24) {
    float s = 0.f;
    for (int nt = 0; nt < 16; ++nt) s += part[i * 16 + nt];
    pooled[i] = s * (1.0f / 262144.0f);
  }
}

__global__ void k_cls(const float* __restrict__ pooled, const float* __restrict__ clsW,
                      const float* __restrict__ clsb, float* __restrict__ out) {
  int i = blockIdx.x * 256 + threadIdx.x;
  if (i < 8 * NCLS) {
    int b = i / NCLS, n = i % NCLS;
    float s = clsb[n];
#pragma unroll
    for (int c = 0; c < 3; ++c) s += pooled[b * 3 + c] * clsW[n * 3 + c];
    out[i] = s;
  }
}

extern "C" void kernel_launch(void* const* d_in, const int* in_sizes, int n_in,
                              void* d_out, int out_size, void* d_ws, size_t ws_size,
                              hipStream_t stream) {
  (void)in_sizes; (void)n_in; (void)out_size; (void)ws_size;
  const float* x    = (const float*)d_in[0];
  const float* Wg   = (const float*)d_in[1];
  const float* Wu   = (const float*)d_in[2];
  const float* Wd   = (const float*)d_in[3];
  const float* ln_w = (const float*)d_in[4];
  const float* clsW = (const float*)d_in[5];
  const float* clsb = (const float*)d_in[6];
  float* out = (float*)d_out;

  char* ws = (char*)d_ws;
  signed char* wq     = (signed char*)(ws);
  signed char* xq     = (signed char*)(ws + 50331648);
  signed char* hq     = (signed char*)(ws + 56623104);
  float*       tbuf   = (float*)      (ws + 81788928);
  float*       xinv   = (float*)      (ws + 182452224);
  float*       hinv   = (float*)      (ws + 182464512);
  float*       wpart  = (float*)      (ws + 182476800);
  float*       winv   = (float*)      (ws + 182489088);
  float*       part   = (float*)      (ws + 182489344);
  float*       pooled = (float*)      (ws + 182495488);

  signed char* wqg = wq;
  signed char* wqu = wq + WN;
  signed char* wqd = wq + 2 * WN;

  k_wabs_part<<<3072, 256, 0, stream>>>(Wg, Wu, Wd, wpart);
  k_wscale   <<<3, 256, 0, stream>>>(wpart, winv);
  k_wquant   <<<49152, 256, 0, stream>>>(Wg, Wu, Wd, winv, wq);
  k_actq     <<<TOK, 256, 0, stream>>>(x, xq, xinv);
  k_gemm_gu  <<<dim3(TOK / 128, FF / 128), 256, 0, stream>>>(xq, wqg, wqu, xinv, winv, tbuf);
  k_rms_quant<<<TOK, 256, 0, stream>>>(tbuf, ln_w, hq, hinv);
  k_gemm_down<<<dim3(TOK / 128, DD / 128), 256, 0, stream>>>(hq, wqd, hinv, winv, part);
  k_pool     <<<1, 32, 0, stream>>>(part, pooled);
  k_cls      <<<32, 256, 0, stream>>>(pooled, clsW, clsb, out);
}
